// LSTM_80169859547569
// MI455X (gfx1250) — matrix operand order repaired and checked by reading, not compiled, not run on hardware
//
#include <hip/hip_runtime.h>

typedef __attribute__((ext_vector_type(16))) _Float16 v16h;
typedef __attribute__((ext_vector_type(8)))  float    v8f;

#define T_STEPS 40
#define F_IN    92
#define H_DIM   64
#define N4H     256   // 4*H

// Branch-free fast activations: single v_exp_f32 + v_rcp_f32 (TRANS pipe,
// co-executes with WMMA). No libm special-case branches in the hot loop.
__device__ __forceinline__ float sigmoid_fast(float v) {
    return __builtin_amdgcn_rcpf(1.0f + __expf(-v));
}
__device__ __forceinline__ float tanh_fast(float v) {
    // tanh(x) = 2*sigmoid(2x) - 1
    return __builtin_fmaf(2.0f, __builtin_amdgcn_rcpf(1.0f + __expf(-2.0f * v)), -1.0f);
}

__global__ __launch_bounds__(128) void lstm_fused_wmma(
    const float* __restrict__ x,    // [B, T, F]
    const float* __restrict__ Wk,   // [F, 4H]
    const float* __restrict__ Wr,   // [H, 4H]
    const float* __restrict__ b,    // [4H]
    const float* __restrict__ Wd,   // [H, 2]
    const float* __restrict__ bd,   // [2]
    float* __restrict__ out)        // [B, 2]
{
    __shared__ _Float16 h_lds[16 * H_DIM];   // 2 KB: full h tile, row-major [row][hcol]

    const int lane = threadIdx.x & 31;
    const int wave = threadIdx.x >> 5;       // 0..3 -> owns hidden cols [wave*16, wave*16+16)
    const int g    = lane >> 4;              // lane half-group
    const int m    = lane & 15;              // A-row / B-col / D-col index
    const int b0   = blockIdx.x * 16;        // batch tile base

    // ---------------- Load weight B-fragments into registers (reused 40x) ----------------
    // B-layout (16x16x32 f16): lane holds col n = m; elem i -> k_local = (i>=8?16:0) + g*8 + (i&7).
    // Bias trick: Wk row k==F_IN holds b[n]; x fragment supplies 1.0 at k==F_IN.
    v16h wk[3][4];   // [K-chunk][gate], K padded 92 -> 96 (row 92 = bias)
    #pragma unroll
    for (int kc = 0; kc < 3; ++kc) {
        #pragma unroll
        for (int gt = 0; gt < 4; ++gt) {
            const int n = gt * H_DIM + wave * 16 + m;
            v16h tmp;
            #pragma unroll
            for (int i = 0; i < 16; ++i) {
                const int k = kc * 32 + ((i >= 8) ? 16 : 0) + g * 8 + (i & 7);
                float wv;
                if (k < F_IN)       wv = Wk[k * N4H + n];
                else if (k == F_IN) wv = b[n];          // bias row
                else                wv = 0.0f;
                tmp[i] = (_Float16)wv;
            }
            wk[kc][gt] = tmp;
        }
    }
    v16h wr[2][4];   // K = 64 exactly
    #pragma unroll
    for (int kc = 0; kc < 2; ++kc) {
        #pragma unroll
        for (int gt = 0; gt < 4; ++gt) {
            const int n = gt * H_DIM + wave * 16 + m;
            v16h tmp;
            #pragma unroll
            for (int i = 0; i < 16; ++i) {
                const int k = kc * 32 + ((i >= 8) ? 16 : 0) + g * 8 + (i & 7);
                tmp[i] = (_Float16)Wr[k * N4H + n];
            }
            wr[kc][gt] = tmp;
        }
    }

    // ---------------- Recurrent state ----------------
    v8f c_st;
    #pragma unroll
    for (int r = 0; r < 8; ++r) c_st[r] = 0.0f;

    v16h a_h[2];                              // h in A-layout (h0 = 0)
    #pragma unroll
    for (int i = 0; i < 16; ++i) { a_h[0][i] = (_Float16)0.0f; a_h[1][i] = (_Float16)0.0f; }

    const float* xrow = x + (size_t)(b0 + m) * (T_STEPS * F_IN);

    // ---- Raw x quad loads (b128), one-deep software pipeline ----
    // Fragment elem i of tile kc has k = kc*32 + (i>=8?16:0) + g*8 + (i&7);
    // quad j (i = 4j..4j+3) starts at kb = kc*32 + (j>=2?16:0) + g*8 + (j&1)*4.
    // Only kb==92 (kc=2,j=3,g=1) is out of range -> bias quad {1,0,0,0}.
    float4 xraw[12];
    auto load_xraw = [&](int t, float4* q) {
        const float* xp = xrow + t * F_IN;
        #pragma unroll
        for (int kc = 0; kc < 3; ++kc) {
            #pragma unroll
            for (int j = 0; j < 4; ++j) {
                const int kb = kc * 32 + ((j >= 2) ? 16 : 0) + g * 8 + (j & 1) * 4;
                if (kb < F_IN)
                    q[kc * 4 + j] = *reinterpret_cast<const float4*>(xp + kb);
                else
                    q[kc * 4 + j] = make_float4(1.0f, 0.0f, 0.0f, 0.0f);  // bias one-hot
            }
        }
    };
    load_xraw(0, xraw);

    for (int t = 0; t < T_STEPS; ++t) {
        // ---- Convert this step's raw quads to A-fragments (loads issued last iter) ----
        v16h a_x[3];
        #pragma unroll
        for (int kc = 0; kc < 3; ++kc) {
            #pragma unroll
            for (int j = 0; j < 4; ++j) {
                const float4 q = xraw[kc * 4 + j];
                a_x[kc][4 * j + 0] = (_Float16)q.x;
                a_x[kc][4 * j + 1] = (_Float16)q.y;
                a_x[kc][4 * j + 2] = (_Float16)q.z;
                a_x[kc][4 * j + 3] = (_Float16)q.w;
            }
        }
        // ---- Issue next step's loads early (branchless: clamp to last step) ----
        const int tn = (t + 1 < T_STEPS) ? (t + 1) : (T_STEPS - 1);
        load_xraw(tn, xraw);

        // ---- z = x@Wk(+bias row) + h@Wr via WMMA (gate gt = N-tile wave + 4*gt) ----
        v8f acc[4] = {};   // zero C start; bias arrives via K-padding
        #pragma unroll
        for (int gt = 0; gt < 4; ++gt) {
            #pragma unroll
            for (int kc = 0; kc < 3; ++kc)
                acc[gt] = __builtin_amdgcn_wmma_f32_16x16x32_f16(
                    false, a_x[kc], false, wk[kc][gt], (short)0, acc[gt], false, false);
            #pragma unroll
            for (int kc = 0; kc < 2; ++kc)
                acc[gt] = __builtin_amdgcn_wmma_f32_16x16x32_f16(
                    false, a_h[kc], false, wr[kc][gt], (short)0, acc[gt], false, false);
        }

        // ---- Elementwise cell update (D-layout: 8 rows per lane), branch-free ----
        v8f hnew;
        #pragma unroll
        for (int r = 0; r < 8; ++r) {
            const float iv = sigmoid_fast(acc[0][r]);
            const float fv = sigmoid_fast(acc[1][r]);
            const float gv = tanh_fast(acc[2][r]);
            const float ov = sigmoid_fast(acc[3][r]);
            const float cv = __builtin_fmaf(fv, c_st[r], iv * gv);
            c_st[r] = cv;
            hnew[r] = ov * tanh_fast(cv);
        }

        // ---- Re-stripe h: D-layout -> row-major LDS -> A-layout fragments ----
        __syncthreads();   // previous iteration's a_h reads complete everywhere
        #pragma unroll
        for (int r = 0; r < 8; ++r) {
            const int row = r + g * 8;            // D-layout row for this lane/VGPR
            const int col = wave * 16 + m;        // this wave's hidden columns
            h_lds[row * H_DIM + col] = (_Float16)hnew[r];
        }
        __syncthreads();
        #pragma unroll
        for (int kc = 0; kc < 2; ++kc) {
            #pragma unroll
            for (int i = 0; i < 16; ++i) {
                const int k = kc * 32 + ((i >= 8) ? 16 : 0) + g * 8 + (i & 7);
                a_h[kc][i] = h_lds[m * H_DIM + k];   // contiguous 16B runs -> ds_load_b128
            }
        }
    }

    // ---------------- Dense head: out = sigmoid(h @ Wd + bd), [16 x 2] per block ----------------
    if (threadIdx.x < 32) {
        const int row   = threadIdx.x & 15;
        const int label = threadIdx.x >> 4;
        float sum = bd[label];
        #pragma unroll
        for (int k = 0; k < H_DIM; ++k)
            sum += (float)h_lds[row * H_DIM + k] * Wd[k * 2 + label];
        out[(size_t)(b0 + row) * 2 + label] = sigmoid_fast(sum);
    }
}

extern "C" void kernel_launch(void* const* d_in, const int* in_sizes, int n_in,
                              void* d_out, int out_size, void* d_ws, size_t ws_size,
                              hipStream_t stream) {
    const float* x  = (const float*)d_in[0];
    const float* Wk = (const float*)d_in[1];
    const float* Wr = (const float*)d_in[2];
    const float* b  = (const float*)d_in[3];
    const float* Wd = (const float*)d_in[4];
    const float* bd = (const float*)d_in[5];
    float* out = (float*)d_out;

    const int blocks = 32768 / 16;   // one 16-row batch tile per block
    lstm_fused_wmma<<<blocks, 128, 0, stream>>>(x, Wk, Wr, b, Wd, bd, out);
}
